// FullModel_4449586118673
// MI455X (gfx1250) — matrix operand order repaired and checked by reading, not compiled, not run on hardware
//
#include <hip/hip_runtime.h>

typedef __bf16 bf16;
typedef __attribute__((ext_vector_type(16))) __bf16 v16bf;
typedef __attribute__((ext_vector_type(8)))  __bf16 v8bf;
typedef __attribute__((ext_vector_type(8)))  float  v8f;

static constexpr int N = 50000, M = 10000, E = 200000;
static constexpr int IN_CH = 384, HID = 128, OUT = 64, HEADS = 4;
static constexpr int HD4 = HID * HEADS; // 512

#define DEVFN static __device__ __forceinline__

DEVFN float warp_sum(float v) {
#pragma unroll
  for (int o = 16; o > 0; o >>= 1) v += __shfl_xor(v, o, 32);
  return v;
}

// order-preserving float <-> uint mapping for atomic min/max on floats
DEVFN unsigned fmap(float f) {
  unsigned u = __float_as_uint(f);
  return (u >> 31) ? ~u : (u | 0x80000000u);
}
DEVFN float funmap(unsigned u) {
  return (u >> 31) ? __uint_as_float(u & 0x7FFFFFFFu) : __uint_as_float(~u);
}

DEVFN float leaky01(float a) { return a > 0.f ? a : 0.01f * a; }
DEVFN float leaky2(float a)  { return a > 0.f ? a : 0.2f  * a; }

DEVFN unsigned pack_bf16(float lo, float hi) {
  unsigned short l = __builtin_bit_cast(unsigned short, (bf16)lo);
  unsigned short h = __builtin_bit_cast(unsigned short, (bf16)hi);
  return (unsigned)l | ((unsigned)h << 16);
}

// CDNA5 packed bf16 atomic add (no HIP builtin) — RMW resolves in L2, DEV scope
DEVFN void atomic_pk_add_bf16(bf16* addr, unsigned packed) {
  asm volatile("global_atomic_pk_add_bf16 %0, %1, off scope:SCOPE_DEV"
               :
               : "v"((unsigned long long)(size_t)addr), "v"(packed)
               : "memory");
}

// ---------------- elementwise f32 -> bf16 ----------------
__global__ void k_f2bf(const float* __restrict__ in, bf16* __restrict__ out, long long n) {
  long long i = (long long)blockIdx.x * blockDim.x + threadIdx.x;
  if (i < n) out[i] = (bf16)in[i];
}

// ---------------- fused l2-normalize + LayerNorm -> bf16, K=384 ----------------
__global__ void k_l2ln_bf16(const float* __restrict__ X, const float* __restrict__ g,
                            const float* __restrict__ b, bf16* __restrict__ Y, int K) {
  int row = blockIdx.x;
  int tid = threadIdx.x;
  const float* xr = X + (size_t)row * K;
  __shared__ float sh[8];
  int lane = tid & 31, w = tid >> 5, nw = blockDim.x >> 5;

  float ss = 0.f;
  for (int i = tid; i < K; i += blockDim.x) { float v = xr[i]; ss += v * v; }
  ss = warp_sum(ss);
  if (lane == 0) sh[w] = ss;
  __syncthreads();
  float tot = 0.f;
  for (int i = 0; i < nw; i++) tot += sh[i];
  float inv = 1.f / fmaxf(sqrtf(tot), 1e-12f);
  __syncthreads();

  float sm = 0.f;
  for (int i = tid; i < K; i += blockDim.x) sm += xr[i] * inv;
  sm = warp_sum(sm);
  if (lane == 0) sh[w] = sm;
  __syncthreads();
  tot = 0.f;
  for (int i = 0; i < nw; i++) tot += sh[i];
  float mu = tot / K;
  __syncthreads();

  float vs = 0.f;
  for (int i = tid; i < K; i += blockDim.x) { float d = xr[i] * inv - mu; vs += d * d; }
  vs = warp_sum(vs);
  if (lane == 0) sh[w] = vs;
  __syncthreads();
  tot = 0.f;
  for (int i = 0; i < nw; i++) tot += sh[i];
  float rstd = rsqrtf(tot / K + 1e-5f);

  for (int i = tid; i < K; i += blockDim.x)
    Y[(size_t)row * K + i] = (bf16)((xr[i] * inv - mu) * rstd * g[i] + b[i]);
}

// ---------------- LayerNorm over 128 -> bf16, one wave per row ----------------
__global__ void k_ln128_bf16(const float* __restrict__ X, const float* __restrict__ g,
                             const float* __restrict__ b, bf16* __restrict__ Y, int rows) {
  int wid = blockIdx.x * (blockDim.x >> 5) + (threadIdx.x >> 5);
  int lane = threadIdx.x & 31;
  if (wid >= rows) return;
  const float* xr = X + (size_t)wid * HID;
  float x0 = xr[lane], x1 = xr[lane + 32], x2 = xr[lane + 64], x3 = xr[lane + 96];
  float mu = warp_sum(x0 + x1 + x2 + x3) * (1.f / HID);
  float d0 = x0 - mu, d1 = x1 - mu, d2 = x2 - mu, d3 = x3 - mu;
  float var = warp_sum(d0 * d0 + d1 * d1 + d2 * d2 + d3 * d3) * (1.f / HID);
  float rstd = rsqrtf(var + 1e-5f);
  bf16* yr = Y + (size_t)wid * HID;
  yr[lane]      = (bf16)(d0 * rstd * g[lane]      + b[lane]);
  yr[lane + 32] = (bf16)(d1 * rstd * g[lane + 32] + b[lane + 32]);
  yr[lane + 64] = (bf16)(d2 * rstd * g[lane + 64] + b[lane + 64]);
  yr[lane + 96] = (bf16)(d3 * rstd * g[lane + 96] + b[lane + 96]);
}

// ---------------- WMMA GEMM: C[rows,ncols] = A[rows,K](bf16) @ W[ncols,K]^T ----------------
// one wave per 16x64 output tile (4 column sub-tiles share one A fragment);
// rows%16==0, ncols%64==0, K%32==0
__global__ void k_gemm_bf16(const bf16* __restrict__ A, const bf16* __restrict__ W,
                            const float* __restrict__ bias, float* __restrict__ Cf,
                            bf16* __restrict__ Cb, int K, int ncols, int act) {
  int lane = threadIdx.x;
  int l15 = lane & 15;
  int half = lane >> 4;
  int rowBase = blockIdx.x << 4;
  int colBase = blockIdx.y << 6;
  // A layout (16-bit 16x32): lanes0-15 row=l15 hold K 0..15 contiguous; lanes16-31 hold K 16..31
  const bf16* arow = A + (size_t)(rowBase + l15) * K + half * 16;
  // B layout (16-bit 32x16): lanes0-15 col=l15 hold K 0..15 contiguous; lanes16-31 hold K 16..31
  const bf16* wrow = W + (size_t)(colBase + l15) * K + half * 16;
  size_t wstep = (size_t)16 * K; // 16 output columns
  v8f c[4] = {v8f{}, v8f{}, v8f{}, v8f{}};
  for (int k0 = 0; k0 < K; k0 += 32) {
    v8bf a0 = *(const v8bf*)(arow + k0);
    v8bf a1 = *(const v8bf*)(arow + k0 + 8);
    v16bf a;
#pragma unroll
    for (int i = 0; i < 8; i++) { a[i] = a0[i]; a[i + 8] = a1[i]; }
    v16bf b0 = *(const v16bf*)(wrow + k0);
    v16bf b1 = *(const v16bf*)(wrow + wstep + k0);
    v16bf b2 = *(const v16bf*)(wrow + 2 * wstep + k0);
    v16bf b3 = *(const v16bf*)(wrow + 3 * wstep + k0);
    c[0] = __builtin_amdgcn_wmma_f32_16x16x32_bf16(false, a, false, b0, (short)0, c[0], false, false);
    c[1] = __builtin_amdgcn_wmma_f32_16x16x32_bf16(false, a, false, b1, (short)0, c[1], false, false);
    c[2] = __builtin_amdgcn_wmma_f32_16x16x32_bf16(false, a, false, b2, (short)0, c[2], false, false);
    c[3] = __builtin_amdgcn_wmma_f32_16x16x32_bf16(false, a, false, b3, (short)0, c[3], false, false);
  }
#pragma unroll
  for (int t = 0; t < 4; t++) {
    int ccol = colBase + t * 16 + l15;
    float bv = bias ? bias[ccol] : 0.f;
#pragma unroll
    for (int v = 0; v < 8; v++) {
      int crow = rowBase + v + half * 8;
      float val = c[t][v] + bv;
      if (act == 1) val = leaky01(val);
      size_t offn = (size_t)crow * ncols + ccol;
      if (Cf) Cf[offn] = val;
      else    Cb[offn] = (bf16)val;
    }
  }
}

// ---------------- degree counts + reciprocal ----------------
__global__ void k_deg(const int* __restrict__ row, const int* __restrict__ col,
                      float* __restrict__ dc, float* __restrict__ bc, int e_cnt) {
  int e = blockIdx.x * blockDim.x + threadIdx.x;
  if (e < e_cnt) {
    atomicAdd(&dc[row[e]], 1.f);
    atomicAdd(&bc[col[e]], 1.f);
  }
}
__global__ void k_recip(float* __restrict__ d, int n) {
  int i = blockIdx.x * blockDim.x + threadIdx.x;
  if (i < n) d[i] = d[i] > 0.f ? 1.f / d[i] : 0.f;
}

// ---------------- generic scatter-add: dst[didx[e]] += scale(sidx[e]) * src[sidx[e]] ----------------
__global__ void k_scatter_add(const float* __restrict__ src, const int* __restrict__ sidx,
                              const int* __restrict__ didx, const float* __restrict__ sscale,
                              float* __restrict__ dst, int shift, int e_cnt) {
  long long t = (long long)blockIdx.x * blockDim.x + threadIdx.x;
  int e = (int)(t >> shift);
  if (e >= e_cnt) return;
  int d = (int)(t & ((1 << shift) - 1));
  int si = sidx[e];
  float v = src[((size_t)si << shift) + d];
  if (sscale) v *= sscale[si];
  atomicAdd(&dst[((size_t)didx[e] << shift) + d], v);
}

// ---------------- hgconv finalize: acc = leaky01(Dn*acc + bias) ----------------
__global__ void k_conv_fin(float* __restrict__ acc, const float* __restrict__ Dn,
                           const float* __restrict__ bias, long long n) {
  long long t = (long long)blockIdx.x * blockDim.x + threadIdx.x;
  if (t >= n) return;
  int i = (int)(t >> 7), d = (int)(t & 127);
  acc[t] = leaky01(Dn[i] * acc[t] + bias[d]);
}

// ---------------- per-(row,head) attention scores: dot over 128 ----------------
__global__ void k_att_scores(const bf16* __restrict__ xw, const float* __restrict__ att,
                             float* __restrict__ sc, int attOff) {
  int i = blockIdx.x;
  int h = threadIdx.x >> 5;
  int lane = threadIdx.x & 31;
  const bf16* xr = xw + (size_t)i * HD4 + h * HID;
  const float* ar = att + h * (2 * HID) + attOff;
  float s = (float)xr[lane] * ar[lane] + (float)xr[lane + 32] * ar[lane + 32] +
            (float)xr[lane + 64] * ar[lane + 64] + (float)xr[lane + 96] * ar[lane + 96];
  s = warp_sum(s);
  if (lane == 0) sc[(size_t)i * HEADS + h] = s;
}

// ---------------- segment softmax over col ----------------
__global__ void k_att_max(const int* __restrict__ row, const int* __restrict__ col,
                          const float* __restrict__ an, const float* __restrict__ ee,
                          unsigned* __restrict__ amax, int e_cnt) {
  int t = blockIdx.x * blockDim.x + threadIdx.x;
  int e = t >> 2, h = t & 3;
  if (e >= e_cnt) return;
  float a = leaky2(an[(size_t)row[e] * 4 + h] + ee[(size_t)col[e] * 4 + h]);
  atomicMax(&amax[(size_t)col[e] * 4 + h], fmap(a));
}
__global__ void k_att_exp(const int* __restrict__ row, const int* __restrict__ col,
                          const float* __restrict__ an, const float* __restrict__ ee,
                          const unsigned* __restrict__ amax, float* __restrict__ expa,
                          float* __restrict__ ssum, int e_cnt) {
  int t = blockIdx.x * blockDim.x + threadIdx.x;
  int e = t >> 2, h = t & 3;
  if (e >= e_cnt) return;
  int ci = col[e];
  float a = leaky2(an[(size_t)row[e] * 4 + h] + ee[(size_t)ci * 4 + h]);
  float ex = __expf(a - funmap(amax[(size_t)ci * 4 + h]));
  expa[(size_t)e * 4 + h] = ex;
  atomicAdd(&ssum[(size_t)ci * 4 + h], ex);
}
__global__ void k_att_alpha(const int* __restrict__ col, const float* __restrict__ ssum,
                            float* __restrict__ expa, int e_cnt) {
  int t = blockIdx.x * blockDim.x + threadIdx.x;
  int e = t >> 2, h = t & 3;
  if (e >= e_cnt) return;
  expa[t] = expa[t] / (ssum[(size_t)col[e] * 4 + h] + 1e-16f);
}

// ---------------- attention message passes (packed-bf16 atomics) ----------------
// m[col] += alpha * xw[row]    (m is bf16 [M,512], pk_add_bf16 atomics)
__global__ void k_att_m(const bf16* __restrict__ xw, const int* __restrict__ row,
                        const int* __restrict__ col, const float* __restrict__ alpha,
                        bf16* __restrict__ m, int e_cnt) {
  long long t = (long long)blockIdx.x * blockDim.x + threadIdx.x;
  int e = (int)(t >> 8);
  if (e >= e_cnt) return;
  int d = (int)(t & 255) * 2; // channel pair
  int h = d >> 7;
  float al = alpha[(size_t)e * 4 + h];
  const bf16* xp = xw + (size_t)row[e] * HD4 + d;
  unsigned pk = pack_bf16((float)xp[0] * al, (float)xp[1] * al);
  atomic_pk_add_bf16(m + (size_t)col[e] * HD4 + d, pk);
}
// acc[row] += alpha * Bn[col] * m[col]   (f32 accumulation, heads summed in place)
__global__ void k_att_out(const bf16* __restrict__ m, const int* __restrict__ row,
                          const int* __restrict__ col, const float* __restrict__ alpha,
                          const float* __restrict__ Bn, float* __restrict__ acc, int e_cnt) {
  long long t = (long long)blockIdx.x * blockDim.x + threadIdx.x;
  int e = (int)(t >> 8);
  if (e >= e_cnt) return;
  int d = (int)(t & 255) * 2;
  int h = d >> 7, dd = d & 127;
  int ci = col[e];
  float s = Bn[ci] * alpha[(size_t)e * 4 + h];
  const bf16* mp = m + (size_t)ci * HD4 + d;
  float* ap = acc + (size_t)row[e] * HID + dd;
  atomicAdd(ap,     (float)mp[0] * s);
  atomicAdd(ap + 1, (float)mp[1] * s);
}
// x_final = leaky01(Dn*acc/4 + bias) + hs  (residual with struct branch)
__global__ void k_att_fin(const float* __restrict__ acc, const float* __restrict__ Dn,
                          const float* __restrict__ bias, const float* __restrict__ hs,
                          float* __restrict__ xfin, long long n) {
  long long t = (long long)blockIdx.x * blockDim.x + threadIdx.x;
  if (t >= n) return;
  int i = (int)(t >> 7), d = (int)(t & 127);
  xfin[t] = leaky01(Dn[i] * acc[t] * 0.25f + bias[d]) + hs[t];
}

// ---------------- segment-min over col ----------------
__global__ void k_segmin(const int* __restrict__ row, const int* __restrict__ col,
                         const float* __restrict__ xf, unsigned* __restrict__ agg, int e_cnt) {
  long long t = (long long)blockIdx.x * blockDim.x + threadIdx.x;
  int e = (int)(t >> 7);
  if (e >= e_cnt) return;
  int d = (int)(t & 127);
  atomicMin(&agg[(size_t)col[e] * HID + d], fmap(xf[(size_t)row[e] * HID + d]));
}
__global__ void k_u2bf(const unsigned* __restrict__ agg, bf16* __restrict__ out, long long n) {
  long long i = (long long)blockIdx.x * blockDim.x + threadIdx.x;
  if (i < n) out[i] = (bf16)funmap(agg[i]);
}

// =========================================================================
extern "C" void kernel_launch(void* const* d_in, const int* in_sizes, int n_in,
                              void* d_out, int out_size, void* d_ws, size_t ws_size,
                              hipStream_t stream) {
  (void)in_sizes; (void)n_in; (void)out_size; (void)ws_size;
  const float* x        = (const float*)d_in[0];
  const float* x_e      = (const float*)d_in[1];
  const int*   ei       = (const int*)d_in[2];
  const float* xstruct  = (const float*)d_in[3];
  const float* sem_g    = (const float*)d_in[4];
  const float* sem_b    = (const float*)d_in[5];
  const float* sem_w    = (const float*)d_in[6];
  const float* sem_pb   = (const float*)d_in[7];
  const float* eg       = (const float*)d_in[8];
  const float* eb       = (const float*)d_in[9];
  const float* e_w      = (const float*)d_in[10];
  const float* e_pb     = (const float*)d_in[11];
  const float* in_g     = (const float*)d_in[12];
  const float* in_b     = (const float*)d_in[13];
  const float* in_w     = (const float*)d_in[14];
  const float* in_pb    = (const float*)d_in[15];
  const float* n0_g     = (const float*)d_in[16];
  const float* n0_b     = (const float*)d_in[17];
  const float* hg0_w    = (const float*)d_in[18];
  const float* hg0_bias = (const float*)d_in[19];
  const float* n0l_g    = (const float*)d_in[20];
  const float* n0l_b    = (const float*)d_in[21];
  const float* hg0l_w   = (const float*)d_in[22];
  const float* hg0l_att = (const float*)d_in[23];
  const float* hg0l_bias= (const float*)d_in[24];
  const float* lin_w    = (const float*)d_in[25];
  const float* lin_b    = (const float*)d_in[26];
  const int* row = ei;
  const int* col = ei + E;
  float* outp = (float*)d_out;

  char* ws = (char*)d_ws;
  size_t off = 0;
  auto alloc = [&](size_t bytes) -> char* {
    char* p = ws + off;
    off += (bytes + 255) & ~(size_t)255;
    return p;
  };

  // bf16 weight copies
  bf16* wb_sem  = (bf16*)alloc((size_t)HID * IN_CH * 2);
  bf16* wb_e    = (bf16*)alloc((size_t)HID * IN_CH * 2);
  bf16* wb_in   = (bf16*)alloc((size_t)HID * IN_CH * 2);
  bf16* wb_hg0  = (bf16*)alloc((size_t)HID * HID * 2);
  bf16* wb_hg0l = (bf16*)alloc((size_t)HD4 * HID * 2);
  bf16* wb_lin  = (bf16*)alloc((size_t)OUT * HID * 2);
  // activations / intermediates
  bf16*  xnbf  = (bf16*)alloc((size_t)N * IN_CH * 2);   // normalized x / x_struct (bf16)
  bf16*  xebf  = (bf16*)alloc((size_t)M * IN_CH * 2);   // normalized x_e (bf16)
  float* h     = (float*)alloc((size_t)N * HID * 4);    // semantic branch; later x_final
  float* he    = (float*)alloc((size_t)M * HID * 4);    // edge features
  float* hs    = (float*)alloc((size_t)N * HID * 4);    // struct branch; reused as conv acc
  bf16*  lnbf  = (bf16*)alloc((size_t)N * HID * 2);     // ln outputs (bf16)
  bf16*  hebf  = (bf16*)alloc((size_t)M * HID * 2);
  float* xw1   = (float*)alloc((size_t)N * HID * 4);    // hgconv xw; reused as att acc
  float* mbuf  = (float*)alloc((size_t)M * HD4 * 4);    // hgconv messages (f32)
  bf16*  mpk   = (bf16*)mbuf;                           // attention messages (bf16 alias)
  bf16*  xw2bf = (bf16*)alloc((size_t)N * HD4 * 2);     // attention xw (bf16)
  bf16*  ewbf  = (bf16*)alloc((size_t)M * HD4 * 2);
  float* an    = (float*)alloc((size_t)N * HEADS * 4);
  float* ees   = (float*)alloc((size_t)M * HEADS * 4);
  unsigned* amaxu = (unsigned*)alloc((size_t)M * HEADS * 4);
  float* ssum  = (float*)alloc((size_t)M * HEADS * 4);
  float* expa  = (float*)alloc((size_t)E * HEADS * 4);  // becomes alpha in place
  float* Dn    = (float*)alloc((size_t)N * 4);
  float* Bn    = (float*)alloc((size_t)M * 4);
  unsigned* aggu = (unsigned*)alloc((size_t)M * HID * 4);
  bf16*  aggbf = (bf16*)alloc((size_t)M * HID * 2);

  auto cdiv = [](long long a, long long b) { return (int)((a + b - 1) / b); };

  // ---- weights -> bf16 ----
  k_f2bf<<<cdiv((long long)HID * IN_CH, 256), 256, 0, stream>>>(sem_w,  wb_sem,  (long long)HID * IN_CH);
  k_f2bf<<<cdiv((long long)HID * IN_CH, 256), 256, 0, stream>>>(e_w,    wb_e,    (long long)HID * IN_CH);
  k_f2bf<<<cdiv((long long)HID * IN_CH, 256), 256, 0, stream>>>(in_w,   wb_in,   (long long)HID * IN_CH);
  k_f2bf<<<cdiv((long long)HID * HID, 256), 256, 0, stream>>>(hg0_w,  wb_hg0,  (long long)HID * HID);
  k_f2bf<<<cdiv((long long)HD4 * HID, 256), 256, 0, stream>>>(hg0l_w, wb_hg0l, (long long)HD4 * HID);
  k_f2bf<<<cdiv((long long)OUT * HID, 256), 256, 0, stream>>>(lin_w,  wb_lin,  (long long)OUT * HID);

  // ---- degrees ----
  hipMemsetAsync(Dn, 0, (size_t)N * 4, stream);
  hipMemsetAsync(Bn, 0, (size_t)M * 4, stream);
  k_deg<<<cdiv(E, 256), 256, 0, stream>>>(row, col, Dn, Bn, E);
  k_recip<<<cdiv(N, 256), 256, 0, stream>>>(Dn, N);
  k_recip<<<cdiv(M, 256), 256, 0, stream>>>(Bn, M);

  // ---- semantic branch: x -> h ----
  k_l2ln_bf16<<<N, 128, 0, stream>>>(x, sem_g, sem_b, xnbf, IN_CH);
  k_gemm_bf16<<<dim3(N / 16, HID / 64), 32, 0, stream>>>(xnbf, wb_sem, sem_pb, h, nullptr, IN_CH, HID, 1);

  // ---- edge branch: x_e -> he ----
  k_l2ln_bf16<<<M, 128, 0, stream>>>(x_e, eg, eb, xebf, IN_CH);
  k_gemm_bf16<<<dim3(M / 16, HID / 64), 32, 0, stream>>>(xebf, wb_e, e_pb, he, nullptr, IN_CH, HID, 1);

  // ---- struct branch: x_struct -> hs ----
  k_l2ln_bf16<<<N, 128, 0, stream>>>(xstruct, in_g, in_b, xnbf, IN_CH);
  k_gemm_bf16<<<dim3(N / 16, HID / 64), 32, 0, stream>>>(xnbf, wb_in, in_pb, hs, nullptr, IN_CH, HID, 1);

  // ---- hgconv (struct): ln -> gemm -> two scatter passes -> finalize ----
  k_ln128_bf16<<<cdiv(N, 8), 256, 0, stream>>>(hs, n0_g, n0_b, lnbf, N);
  k_gemm_bf16<<<dim3(N / 16, HID / 64), 32, 0, stream>>>(lnbf, wb_hg0, nullptr, xw1, nullptr, HID, HID, 0);
  hipMemsetAsync(mbuf, 0, (size_t)M * HID * 4, stream);
  k_scatter_add<<<cdiv((long long)E * HID, 256), 256, 0, stream>>>(xw1, row, col, nullptr, mbuf, 7, E);
  hipMemsetAsync(hs, 0, (size_t)N * HID * 4, stream);
  k_scatter_add<<<cdiv((long long)E * HID, 256), 256, 0, stream>>>(mbuf, col, row, Bn, hs, 7, E);
  k_conv_fin<<<cdiv((long long)N * HID, 256), 256, 0, stream>>>(hs, Dn, hg0_bias, (long long)N * HID);

  // ---- attention conv: ln(h) -> xw2, he -> ew ----
  k_ln128_bf16<<<cdiv(N, 8), 256, 0, stream>>>(h, n0l_g, n0l_b, lnbf, N);
  k_gemm_bf16<<<dim3(N / 16, HD4 / 64), 32, 0, stream>>>(lnbf, wb_hg0l, nullptr, nullptr, xw2bf, HID, HD4, 0);
  k_f2bf<<<cdiv((long long)M * HID, 256), 256, 0, stream>>>(he, hebf, (long long)M * HID);
  k_gemm_bf16<<<dim3(M / 16, HD4 / 64), 32, 0, stream>>>(hebf, wb_hg0l, nullptr, nullptr, ewbf, HID, HD4, 0);

  // ---- attention scores + segment softmax over col ----
  k_att_scores<<<N, 128, 0, stream>>>(xw2bf, hg0l_att, an, 0);
  k_att_scores<<<M, 128, 0, stream>>>(ewbf, hg0l_att, ees, HID);
  hipMemsetAsync(amaxu, 0, (size_t)M * HEADS * 4, stream);
  k_att_max<<<cdiv((long long)E * HEADS, 256), 256, 0, stream>>>(row, col, an, ees, amaxu, E);
  hipMemsetAsync(ssum, 0, (size_t)M * HEADS * 4, stream);
  k_att_exp<<<cdiv((long long)E * HEADS, 256), 256, 0, stream>>>(row, col, an, ees, amaxu, expa, ssum, E);
  k_att_alpha<<<cdiv((long long)E * HEADS, 256), 256, 0, stream>>>(col, ssum, expa, E);

  // ---- attention message passing (packed bf16 atomics on m) ----
  hipMemsetAsync(mpk, 0, (size_t)M * HD4 * 2, stream);
  k_att_m<<<cdiv((long long)E * HD4 / 2, 256), 256, 0, stream>>>(xw2bf, row, col, expa, mpk, E);
  hipMemsetAsync(xw1, 0, (size_t)N * HID * 4, stream);
  k_att_out<<<cdiv((long long)E * HD4 / 2, 256), 256, 0, stream>>>(mpk, row, col, expa, Bn, xw1, E);
  k_att_fin<<<cdiv((long long)N * HID, 256), 256, 0, stream>>>(xw1, Dn, hg0l_bias, hs, h, (long long)N * HID);

  // ---- segment-min + final projection ----
  hipMemsetAsync(aggu, 0xFF, (size_t)M * HID * 4, stream);
  k_segmin<<<cdiv((long long)E * HID, 256), 256, 0, stream>>>(row, col, h, aggu, E);
  k_u2bf<<<cdiv((long long)M * HID, 256), 256, 0, stream>>>(aggu, aggbf, (long long)M * HID);
  k_gemm_bf16<<<dim3(M / 16, OUT / 64), 32, 0, stream>>>(aggbf, wb_lin, lin_b, outp, nullptr, HID, OUT, 0);
}
